// Model_26018911879758
// MI455X (gfx1250) — matrix operand order repaired and checked by reading, not compiled, not run on hardware
//
#include <hip/hip_runtime.h>

typedef __attribute__((ext_vector_type(16))) _Float16 v16h;
typedef __attribute__((ext_vector_type(8)))  _Float16 v8h;
typedef __attribute__((ext_vector_type(8)))  float    v8f;

#define NCN 100000   // customers
#define NRN 100000   // recipes
#define HD  64
#define NE  2000000
#define NL  500000

// ---------------- utility kernels ----------------

__global__ __launch_bounds__(256) void zero_kernel(float* __restrict__ p, long n) {
  long i = (long)blockIdx.x * blockDim.x + threadIdx.x;
  long stride = (long)gridDim.x * blockDim.x;
  for (; i < n; i += stride) p[i] = 0.0f;
}

__global__ __launch_bounds__(256) void degree_kernel(const int* __restrict__ dst,
                                                     float* __restrict__ deg, int E) {
  int e = blockIdx.x * blockDim.x + threadIdx.x;
  if (e < E) unsafeAtomicAdd(&deg[dst[e]], 1.0f);
}

// aggr[dst[e]*F + f..f+3] += x[src[e]*F + f..f+3]   (F = 1<<logF), float4 per thread
__global__ __launch_bounds__(256) void scatter_kernel(const float* __restrict__ x,
                                                      const int* __restrict__ src,
                                                      const int* __restrict__ dst,
                                                      float* __restrict__ aggr,
                                                      int E, int logF) {
  unsigned idx = blockIdx.x * blockDim.x + threadIdx.x;
  int logQ = logF - 2;                       // quads per row
  unsigned total = (unsigned)E << logQ;
  if (idx >= total) return;
  unsigned e = idx >> logQ;
  unsigned q = idx & ((1u << logQ) - 1u);
  const float4 v = *(const float4*)(x + ((long)src[e] << logF) + q * 4);
  float* a = aggr + ((long)dst[e] << logF) + q * 4;
  unsafeAtomicAdd(a + 0, v.x);
  unsafeAtomicAdd(a + 1, v.y);
  unsafeAtomicAdd(a + 2, v.z);
  unsafeAtomicAdd(a + 3, v.w);
}

// ---------------- weight pre-pack ----------------
// Pack f32 W[K x 64] into per-lane WMMA-B layout, f16:
//   out[((kk*4 + t)*32 + lane)*16 + i] = W[kk*32 + (lane>>4)*8 + i + (i>=8)*8][16*t + (lane&15)]
__global__ __launch_bounds__(256) void pack_w_kernel(const float* __restrict__ W,
                                                     _Float16* __restrict__ out, int K) {
  int tid = blockIdx.x * blockDim.x + threadIdx.x;
  int total = (K / 32) * 128;
  if (tid >= total) return;
  int lane = tid & 31;
  int t = (tid >> 5) & 3;
  int kk = tid >> 7;
  int half = lane >> 4;
  int m = lane & 15;
  const float* src = W + (long)(kk * 32 + half * 8) * HD + 16 * t + m;
  _Float16* dst = out + (long)tid * 16;
#pragma unroll
  for (int i = 0; i < 16; ++i) dst[i] = (_Float16)src[(i + (i >= 8 ? 8 : 0)) * HD];
}

// ---------------- WMMA operand helpers ----------------
// A tile 16x32 f16 (ISA 7.12.2): lane<16 -> M=lane, K in {0..7,16..23};
// lane>=16 -> M=lane-16, K in {8..15,24..31}.
__device__ inline v16h load_a_f32(const float* __restrict__ rowptr, int kbase,
                                  int half, float scale) {
  v16h a;
  const float* p0 = rowptr + kbase + half * 8;
#pragma unroll
  for (int i = 0; i < 8; ++i) a[i] = (_Float16)(p0[i] * scale);
  const float* p1 = p0 + 16;
#pragma unroll
  for (int i = 0; i < 8; ++i) a[8 + i] = (_Float16)(p1[i] * scale);
  return a;
}

__device__ inline v16h load_a_f16(const _Float16* __restrict__ rowptr, int kbase, int half) {
  const v8h lo = *(const v8h*)(rowptr + kbase + half * 8);
  const v8h hi = *(const v8h*)(rowptr + kbase + half * 8 + 16);
  v16h a;
#pragma unroll
  for (int i = 0; i < 8; ++i) { a[i] = lo[i]; a[8 + i] = hi[i]; }
  return a;
}

__device__ inline v8f zero_v8f() {
  v8f z;
#pragma unroll
  for (int i = 0; i < 8; ++i) z[i] = 0.0f;
  return z;
}

// ---------------- SAGE layer GEMM (compile-time specialized) ----------------
// out[n,:64] = act( (A1[n,:]*inv_deg[n]) @ W1 + A2[n,:] @ W2 + bias )
// B operands pre-packed f16; K1/K2/operand-type/relu are template params so the
// K-loops fully unroll and all B-tile loads can be clause-grouped.
template <int K1, int K2, bool A2H, bool RELU>
__global__ __launch_bounds__(256)
void sage_gemm_kernel(const float* __restrict__ A1, const float* __restrict__ deg,
                      const v16h* __restrict__ B1p,
                      const float* __restrict__ A2f, const _Float16* __restrict__ A2h,
                      const v16h* __restrict__ B2p,
                      const float* __restrict__ bias, float* __restrict__ out,
                      _Float16* __restrict__ out16, int N) {
  int wave = (blockIdx.x * blockDim.x + threadIdx.x) >> 5;
  int lane = threadIdx.x & 31;
  int r0 = wave * 16;
  if (r0 >= N) return;                 // wave-uniform: EXEC all-ones inside
  int half = lane >> 4;
  int m = lane & 15;
  int arow = r0 + m;

  v8f c[4];
#pragma unroll
  for (int t = 0; t < 4; ++t) c[t] = zero_v8f();

  float scale = 1.0f / fmaxf(deg[arow], 1.0f);

  const float* a1 = A1 + (long)arow * K1;
#pragma unroll
  for (int kk = 0; kk < K1; kk += 32) {
    v16h a = load_a_f32(a1, kk, half, scale);
    const v16h* bp = B1p + (kk >> 5) * 128 + lane;
#pragma unroll
    for (int t = 0; t < 4; ++t) {
      c[t] = __builtin_amdgcn_wmma_f32_16x16x32_f16(false, a, false, bp[t * 32],
                                                    (short)0, c[t], false, false);
    }
  }

#pragma unroll
  for (int kk = 0; kk < K2; kk += 32) {
    v16h a;
    if constexpr (A2H) {
      a = load_a_f16(A2h + (long)arow * K2, kk, half);
    } else {
      a = load_a_f32(A2f + (long)arow * K2, kk, half, 1.0f);
    }
    const v16h* bp = B2p + (kk >> 5) * 128 + lane;
#pragma unroll
    for (int t = 0; t < 4; ++t) {
      c[t] = __builtin_amdgcn_wmma_f32_16x16x32_f16(false, a, false, bp[t * 32],
                                                    (short)0, c[t], false, false);
    }
  }

  // C layout: vgpr j -> row = r0 + half*8 + j, col = 16t + m
#pragma unroll
  for (int t = 0; t < 4; ++t) {
    int col = 16 * t + m;
    float bc = bias[col];
#pragma unroll
    for (int j = 0; j < 8; ++j) {
      float v = c[t][j] + bc;
      if constexpr (RELU) v = fmaxf(v, 0.0f);
      long o = (long)(r0 + half * 8 + j) * HD + col;
      out[o] = v;
      out16[o] = (_Float16)v;
    }
  }
}

// ---------------- decoder ----------------
// per edge l: z = [z_c[row_l], z_r[col_l]] (128 wide)
// h = relu(z @ Wd1 + bd1);  out[l] = h . Wd2 + bd2
__global__ __launch_bounds__(256)
void decoder_kernel(const _Float16* __restrict__ zc16, const _Float16* __restrict__ zr16,
                    const int* __restrict__ eli,
                    const v16h* __restrict__ Wd1p, const float* __restrict__ bd1,
                    const float* __restrict__ Wd2, const float* __restrict__ bd2,
                    float* __restrict__ out, int L) {
  int wave = (blockIdx.x * blockDim.x + threadIdx.x) >> 5;
  int lane = threadIdx.x & 31;
  int l0 = wave * 16;
  if (l0 >= L) return;                 // wave-uniform guard
  int half = lane >> 4;
  int m = lane & 15;
  int edge = l0 + m;
  const _Float16* zcr = zc16 + (long)eli[edge] * HD;
  const _Float16* zrr = zr16 + (long)eli[L + edge] * HD;

  v8f c[4];
#pragma unroll
  for (int t = 0; t < 4; ++t) c[t] = zero_v8f();

#pragma unroll
  for (int kk = 0; kk < 128; kk += 32) {
    const _Float16* arow = (kk < 64) ? (zcr + kk) : (zrr + kk - 64);
    v16h a = load_a_f16(arow, 0, half);
    const v16h* bp = Wd1p + (kk >> 5) * 128 + lane;
#pragma unroll
    for (int t = 0; t < 4; ++t) {
      c[t] = __builtin_amdgcn_wmma_f32_16x16x32_f16(false, a, false, bp[t * 32],
                                                    (short)0, c[t], false, false);
    }
  }

  // second decoder layer: per-lane partials, then reduce over the 16 columns
  float acc[8];
#pragma unroll
  for (int j = 0; j < 8; ++j) acc[j] = 0.0f;
#pragma unroll
  for (int t = 0; t < 4; ++t) {
    int col = 16 * t + m;
    float bc = bd1[col];
    float wc = Wd2[col];
#pragma unroll
    for (int j = 0; j < 8; ++j) acc[j] += fmaxf(c[t][j] + bc, 0.0f) * wc;
  }
#pragma unroll
  for (int mask = 1; mask < 16; mask <<= 1) {
#pragma unroll
    for (int j = 0; j < 8; ++j) acc[j] += __shfl_xor(acc[j], mask, 32);
  }
  if (m == 0) {
    float bb = bd2[0];
#pragma unroll
    for (int j = 0; j < 8; ++j) out[l0 + half * 8 + j] = acc[j] + bb;
  }
}

// ---------------- host launch ----------------

extern "C" void kernel_launch(void* const* d_in, const int* in_sizes, int n_in,
                              void* d_out, int out_size, void* d_ws, size_t ws_size,
                              hipStream_t stream) {
  const float* x_c   = (const float*)d_in[0];
  const float* x_r   = (const float*)d_in[1];
  const float* W1crl = (const float*)d_in[2];
  const float* W1crr = (const float*)d_in[3];
  const float* b1cr  = (const float*)d_in[4];
  const float* W1rcl = (const float*)d_in[5];
  const float* W1rcr = (const float*)d_in[6];
  const float* b1rc  = (const float*)d_in[7];
  const float* W2crl = (const float*)d_in[8];
  const float* W2crr = (const float*)d_in[9];
  const float* b2cr  = (const float*)d_in[10];
  const float* W2rcl = (const float*)d_in[11];
  const float* W2rcr = (const float*)d_in[12];
  const float* b2rc  = (const float*)d_in[13];
  const float* Wd1   = (const float*)d_in[14];
  const float* bd1   = (const float*)d_in[15];
  const float* Wd2   = (const float*)d_in[16];
  const float* bd2   = (const float*)d_in[17];
  const int* ei_c2r  = (const int*)d_in[18];  // [2, E]
  const int* ei_r2c  = (const int*)d_in[19];  // [2, E]
  const int* eli     = (const int*)d_in[20];  // [2, L]
  float* out = (float*)d_out;

  float* ws = (float*)d_ws;
  size_t off = 0;
  float* deg_r = ws + off; off += NRN;
  float* deg_c = ws + off; off += NCN;
  float* bufA  = ws + off; off += (size_t)NRN * 128;  // layer1 aggr(customer), reused layer2
  float* bufB  = ws + off; off += (size_t)NCN * 64;   // layer1 aggr(recipe), reused layer2
  float* h_r   = ws + off; off += (size_t)NRN * HD;
  float* h_c   = ws + off; off += (size_t)NCN * HD;
  float* z_r   = ws + off; off += (size_t)NRN * HD;
  float* z_c   = ws + off; off += (size_t)NCN * HD;

  _Float16* hws = (_Float16*)(ws + off);
  size_t hoff = 0;
  _Float16* h16_r = hws + hoff; hoff += (size_t)NRN * HD;
  _Float16* h16_c = hws + hoff; hoff += (size_t)NCN * HD;
  _Float16* z16_r = hws + hoff; hoff += (size_t)NRN * HD;
  _Float16* z16_c = hws + hoff; hoff += (size_t)NCN * HD;
  // packed weights: (K/32)*4*32*16 halves each
  _Float16* pW1crl = hws + hoff; hoff += 8192;  // K=128
  _Float16* pW1crr = hws + hoff; hoff += 4096;  // K=64
  _Float16* pW1rcl = hws + hoff; hoff += 4096;
  _Float16* pW1rcr = hws + hoff; hoff += 8192;  // K=128
  _Float16* pW2crl = hws + hoff; hoff += 4096;
  _Float16* pW2crr = hws + hoff; hoff += 4096;
  _Float16* pW2rcl = hws + hoff; hoff += 4096;
  _Float16* pW2rcr = hws + hoff; hoff += 4096;
  _Float16* pWd1   = hws + hoff; hoff += 8192;  // K=128
  (void)ws_size; (void)in_sizes; (void)n_in; (void)out_size;

  // ---- pack all weights into WMMA-B f16 layout (one-off, tiny) ----
  pack_w_kernel<<<2, 256, 0, stream>>>(W1crl, pW1crl, 128);
  pack_w_kernel<<<1, 256, 0, stream>>>(W1crr, pW1crr, 64);
  pack_w_kernel<<<1, 256, 0, stream>>>(W1rcl, pW1rcl, 64);
  pack_w_kernel<<<2, 256, 0, stream>>>(W1rcr, pW1rcr, 128);
  pack_w_kernel<<<1, 256, 0, stream>>>(W2crl, pW2crl, 64);
  pack_w_kernel<<<1, 256, 0, stream>>>(W2crr, pW2crr, 64);
  pack_w_kernel<<<1, 256, 0, stream>>>(W2rcl, pW2rcl, 64);
  pack_w_kernel<<<1, 256, 0, stream>>>(W2rcr, pW2rcr, 64);
  pack_w_kernel<<<2, 256, 0, stream>>>(Wd1, pWd1, 128);

  // ---- zero degree + layer-1 accumulators (contiguous region) ----
  long zcount = (long)NRN + NCN + (long)NRN * 128 + (long)NCN * 64;
  zero_kernel<<<8192, 256, 0, stream>>>(deg_r, zcount);

  // ---- degrees (shared by both layers: same edge lists) ----
  degree_kernel<<<(NE + 255) / 256, 256, 0, stream>>>(ei_c2r + NE, deg_r, NE);
  degree_kernel<<<(NE + 255) / 256, 256, 0, stream>>>(ei_r2c + NE, deg_c, NE);

  // ---- layer 1 scatter-add (edge-parallel float4, native f32 atomics, L2-resident) ----
  scatter_kernel<<<(NE * 32u) / 256, 256, 0, stream>>>(x_c, ei_c2r, ei_c2r + NE, bufA, NE, 7);
  scatter_kernel<<<(NE * 16u) / 256, 256, 0, stream>>>(x_r, ei_r2c, ei_r2c + NE, bufB, NE, 6);

  // ---- layer 1 GEMMs (degree normalization folded into A load), relu ----
  int gemm_blocks = ((NRN / 16) + 7) / 8;  // 8 waves/block, 16 rows/wave
  sage_gemm_kernel<128, 64, false, true><<<gemm_blocks, 256, 0, stream>>>(
      bufA, deg_r, (const v16h*)pW1crl, x_r, nullptr, (const v16h*)pW1crr,
      b1cr, h_r, h16_r, NRN);
  sage_gemm_kernel<64, 128, false, true><<<gemm_blocks, 256, 0, stream>>>(
      bufB, deg_c, (const v16h*)pW1rcl, x_c, nullptr, (const v16h*)pW1rcr,
      b1rc, h_c, h16_c, NCN);

  // ---- layer 2: reuse bufA/bufB as 64-wide accumulators ----
  zero_kernel<<<8192, 256, 0, stream>>>(bufA, (long)NRN * 64);
  zero_kernel<<<8192, 256, 0, stream>>>(bufB, (long)NCN * 64);
  scatter_kernel<<<(NE * 16u) / 256, 256, 0, stream>>>(h_c, ei_c2r, ei_c2r + NE, bufA, NE, 6);
  scatter_kernel<<<(NE * 16u) / 256, 256, 0, stream>>>(h_r, ei_r2c, ei_r2c + NE, bufB, NE, 6);

  sage_gemm_kernel<64, 64, true, false><<<gemm_blocks, 256, 0, stream>>>(
      bufA, deg_r, (const v16h*)pW2crl, nullptr, h16_r, (const v16h*)pW2crr,
      b2cr, z_r, z16_r, NRN);
  sage_gemm_kernel<64, 64, true, false><<<gemm_blocks, 256, 0, stream>>>(
      bufB, deg_c, (const v16h*)pW2rcl, nullptr, h16_c, (const v16h*)pW2rcr,
      b2rc, z_c, z16_c, NCN);

  // ---- decoder: 16 edges per wave, WMMA + shuffle-reduce GEMV ----
  int dec_blocks = ((NL / 16) + 7) / 8;
  decoder_kernel<<<dec_blocks, 256, 0, stream>>>(z16_c, z16_r, eli, (const v16h*)pWd1,
                                                 bd1, Wd2, bd2, out, NL);
}
